// GRU_FC_40097814675957
// MI455X (gfx1250) — matrix operand order repaired and checked by reading, not compiled, not run on hardware
//
#include <hip/hip_runtime.h>

#define H      1024
#define B      128
#define SEQ    256
#define NWG    64
#define TPB    256
#define KTILES 32           // K=1024 / 32 per wmma
#define KROUNDS (KTILES/2)  // 2 k-tiles staged per round

typedef unsigned short u16;
typedef unsigned int   u32;
typedef __bf16 bf16;
typedef bf16  v16bf __attribute__((ext_vector_type(16)));
typedef float v8f   __attribute__((ext_vector_type(8)));
typedef int   v4i   __attribute__((ext_vector_type(4)));

#define AS1 __attribute__((address_space(1)))
#define AS3 __attribute__((address_space(3)))

#if defined(__has_builtin)
#  if __has_builtin(__builtin_amdgcn_global_load_async_to_lds_b128)
#    define USE_ASYNC_LDS 1
#  endif
#endif
#ifndef USE_ASYNC_LDS
#  define USE_ASYNC_LDS 0
#endif

struct Pack32 { uint4 lo, hi; };

__device__ __forceinline__ u16 f2bf(float f) {
  u32 u = __builtin_bit_cast(u32, f);
  u32 r = (u + 0x7FFFu + ((u >> 16) & 1u)) >> 16;  // round-to-nearest-even
  return (u16)r;
}
__device__ __forceinline__ float sigm(float x) { return 1.0f / (1.0f + __expf(-x)); }

__device__ __forceinline__ v16bf bits_to_frag(uint4 lo, uint4 hi) {
  Pack32 p{lo, hi};
  return __builtin_bit_cast(v16bf, p);
}

// ---------------- one-shot weight pack: fp32 row-major [3H][H] -> bf16 WMMA-B fragment order
// Packed layout per matrix: tiles (nt,kt), tileId = nt*KTILES+kt, 512 bf16 per tile,
// element (lane,i):  n = nt*16 + (lane&15),  k = kt*32 + (lane>>4)*16 + i   (contiguous k order, same as load_a)
__global__ void pack_weights(const float* __restrict__ whh0, const float* __restrict__ wih1,
                             const float* __restrict__ whh1, const float* __restrict__ h0,
                             u16* __restrict__ wpack, float* __restrict__ hbuf,
                             u16* __restrict__ hbf) {
  const size_t WELE = (size_t)3 * H * H;      // elements per matrix (3H x H)
  const size_t TOTW = 3 * WELE;
  const size_t HE   = (size_t)2 * B * H;      // h0: [L=2][B][H]
  const size_t total = TOTW + HE;
  for (size_t idx = (size_t)blockIdx.x * blockDim.x + threadIdx.x; idx < total;
       idx += (size_t)gridDim.x * blockDim.x) {
    if (idx < TOTW) {
      int    mat = (int)(idx / WELE);
      size_t r   = idx - (size_t)mat * WELE;
      int tile = (int)(r >> 9);
      int lane = (int)((r >> 4) & 31);
      int i    = (int)(r & 15);
      int nt = tile / KTILES;
      int kt = tile % KTILES;
      int n  = nt * 16 + (lane & 15);
      int k  = (kt << 5) + ((lane >> 4) << 4) + i;
      const float* W = (mat == 0) ? whh0 : (mat == 1) ? wih1 : whh1;
      wpack[idx] = f2bf(W[(size_t)n * H + k]);
    } else {
      size_t r = idx - TOTW;                  // parity-0 h buffers get h0
      float v = h0[r];
      hbuf[r] = v;
      hbf[r]  = f2bf(v);
    }
  }
}

// ---------------- persistent GRU ----------------
__device__ __forceinline__ void grid_sync(int* bar) {
  __syncthreads();
  if (threadIdx.x == 0) {
    int g = __hip_atomic_load(&bar[1], __ATOMIC_RELAXED, __HIP_MEMORY_SCOPE_AGENT);
    int a = __hip_atomic_fetch_add(&bar[0], 1, __ATOMIC_ACQ_REL, __HIP_MEMORY_SCOPE_AGENT);
    if (a == NWG - 1) {
      __hip_atomic_store(&bar[0], 0, __ATOMIC_RELAXED, __HIP_MEMORY_SCOPE_AGENT);
      __hip_atomic_fetch_add(&bar[1], 1, __ATOMIC_RELEASE, __HIP_MEMORY_SCOPE_AGENT);
    } else {
      while (__hip_atomic_load(&bar[1], __ATOMIC_ACQUIRE, __HIP_MEMORY_SCOPE_AGENT) == g)
        __builtin_amdgcn_s_sleep(2);
    }
  }
  __syncthreads();
}

// 16-byte memory -> LDS copy; async (ASYNCcnt) on gfx1250 when the builtin exists.
// AS(3) pointer = low 32 bits of the flat shared address (ISA: LDS aperture discards
// the upper bits), AS(1) pointer = the flat global address. Casts go through integers
// so const / address-space qualifiers don't block the conversion; the builtin wants
// exactly (AS1 v4i*, AS3 v4i*, imm, imm) per the round-2 diagnostic.
__device__ __forceinline__ void cp16(uint4* dst, const uint4* src) {
#if USE_ASYNC_LDS
  __builtin_amdgcn_global_load_async_to_lds_b128(
      (AS1 v4i*)(uintptr_t)src,
      (AS3 v4i*)(u32)(uintptr_t)dst, 0, 0);
#else
  *dst = *src;
#endif
}
__device__ __forceinline__ void wait_async() {
#if USE_ASYNC_LDS
#  if __has_builtin(__builtin_amdgcn_s_wait_asynccnt)
  __builtin_amdgcn_s_wait_asynccnt(0);
#  else
  asm volatile("s_wait_asynccnt 0x0" ::: "memory");
#  endif
#endif
}

// stage 2 consecutive k-tiles of the 3 gate B-tiles (6 KB) for column-tile wg into LDS
__device__ __forceinline__ void stage2(uint4* dst /*384 uint4*/, const u16* wmat,
                                       int wg, int rnd, int tid) {
  if (tid < 192) {
    int g = tid >> 6;                         // gate 0..2
    int r = tid & 63;                         // 16B chunk within tile
    size_t t0 = (size_t)((g * (H / 16) + wg) * KTILES + (rnd << 1));
    const uint4* s0 = (const uint4*)(wmat + (t0 << 9));
    cp16(&dst[(g << 7) + r],      s0 + r);       // k-tile 2*rnd
    cp16(&dst[(g << 7) + 64 + r], s0 + 64 + r);  // k-tile 2*rnd+1
  }
}

// A fragment from bf16 row-major [B][H] in contiguous k order:
// lane L: m = wave*16 + (L&15); elems i: K = (L>>4)*16 + i
__device__ __forceinline__ v16bf load_a(const u16* a, int wave, int lane, int kt) {
  int m = (wave << 4) + (lane & 15);
  const u16* p = a + (size_t)m * H + (kt << 5) + ((lane >> 4) << 4);
  const uint4* q = (const uint4*)p;
  return bits_to_frag(q[0], q[1]);            // +0 and +8 bf16
}

__device__ __forceinline__ void gemm3(const u16* __restrict__ wmat, const u16* __restrict__ asrc,
                                      uint4 (*Bsm)[384], int wg, int wave, int lane, int tid,
                                      v8f* acc) {
  stage2(Bsm[0], wmat, wg, 0, tid);
  wait_async();
  __syncthreads();
#pragma unroll 1
  for (int rnd = 0; rnd < KROUNDS; ++rnd) {
    int buf = rnd & 1;
    if (rnd + 1 < KROUNDS) stage2(Bsm[buf ^ 1], wmat, wg, rnd + 1, tid);
#pragma unroll
    for (int sub = 0; sub < 2; ++sub) {
      v16bf af = load_a(asrc, wave, lane, (rnd << 1) + sub);
#pragma unroll
      for (int g = 0; g < 3; ++g) {
        uint4 blo = Bsm[buf][(g << 7) + (sub << 6) + (lane << 1)];
        uint4 bhi = Bsm[buf][(g << 7) + (sub << 6) + (lane << 1) + 1];
        v16bf bfg = bits_to_frag(blo, bhi);
        acc[g] = __builtin_amdgcn_wmma_f32_16x16x32_bf16(false, af, false, bfg,
                                                         (short)0, acc[g], false, false);
      }
    }
    wait_async();                             // staged buf^1 resident before hand-off
    __syncthreads();
  }
}

__global__ void __launch_bounds__(TPB, 1)
gru_persistent(const float* __restrict__ x, const float* __restrict__ wih0,
               const float* __restrict__ bih0, const float* __restrict__ bhh0,
               const float* __restrict__ bih1, const float* __restrict__ bhh1,
               const float* __restrict__ fcw, const float* __restrict__ fcb,
               const int* __restrict__ flen_p,
               const u16* __restrict__ wpack, float* __restrict__ hbuf,
               u16* __restrict__ hbf, float* __restrict__ ybuf, int* bar,
               float* __restrict__ out_mean, float* __restrict__ out_h) {
  __shared__ uint4 Bsm[2][384];               // 12 KB, double-buffered B tiles
  __shared__ float red[TPB];
  __shared__ float s_ysum[2];
  const int tid  = threadIdx.x;
  const int wg   = blockIdx.x;
  const int wave = tid >> 5;
  const int lane = tid & 31;
  const int jbase = wg * 16;                  // 16-column slice of H owned by this WG
  const int F = flen_p[0];
  if (tid < 2) s_ysum[tid] = 0.0f;

  const size_t PH = (size_t)2 * B * H;        // parity stride (covers both layers)
  const size_t LH = (size_t)B * H;            // layer stride
  const u16* Whh0p = wpack;
  const u16* Wih1p = wpack + (size_t)3 * H * H;
  const u16* Whh1p = wpack + (size_t)6 * H * H;

  const v8f vz = {0.f, 0.f, 0.f, 0.f, 0.f, 0.f, 0.f, 0.f};
  int p = 0;                                  // parity holding h_old
  const int total_steps = SEQ + (F - 1);

#pragma unroll 1
  for (int s = 0; s < total_steps; ++s) {
    const bool cond = (s < SEQ);
    // ---------------- layer 0: gh = h0 @ Whh0^T ; gi is scalar broadcast ----------------
    {
      v8f acc[3] = {vz, vz, vz};
      gemm3(Whh0p, hbf + (size_t)p * PH, Bsm, wg, wave, lane, tid, acc);
      const int jn = jbase + (lane & 15);
      const float w_r = wih0[jn], w_z = wih0[H + jn], w_n = wih0[2 * H + jn];
      const float bi_r = bih0[jn], bi_z = bih0[H + jn], bi_n = bih0[2 * H + jn];
      const float bh_r = bhh0[jn], bh_z = bhh0[H + jn], bh_n = bhh0[2 * H + jn];
      const float* hold = hbuf + (size_t)p * PH;
      float* hnf = hbuf + (size_t)(p ^ 1) * PH;
      u16*   hnb = hbf  + (size_t)(p ^ 1) * PH;
#pragma unroll
      for (int v = 0; v < 8; ++v) {
        int m = (wave << 4) + ((lane >> 4) << 3) + v;
        float xv = cond ? x[(size_t)m * SEQ + s] : ybuf[m];
        float ir = xv * w_r + bi_r, iz = xv * w_z + bi_z, in_ = xv * w_n + bi_n;
        float hr = acc[0][v] + bh_r, hz = acc[1][v] + bh_z, hn = acc[2][v] + bh_n;
        float r = sigm(ir + hr), z = sigm(iz + hz);
        float n = tanhf(in_ + r * hn);
        float ho = hold[(size_t)m * H + jn];
        float hv = (1.0f - z) * n + z * ho;
        hnf[(size_t)m * H + jn] = hv;
        hnb[(size_t)m * H + jn] = f2bf(hv);
      }
    }
    grid_sync(bar);
    // ---------------- layer 1: gi = h0_new @ Wih1^T ; gh = h1 @ Whh1^T ----------------
    {
      v8f ai[3] = {vz, vz, vz};
      v8f ah[3] = {vz, vz, vz};
      gemm3(Wih1p, hbf + (size_t)(p ^ 1) * PH, Bsm, wg, wave, lane, tid, ai);
      gemm3(Whh1p, hbf + (size_t)p * PH + LH,  Bsm, wg, wave, lane, tid, ah);
      const int jn = jbase + (lane & 15);
      const float bi_r = bih1[jn], bi_z = bih1[H + jn], bi_n = bih1[2 * H + jn];
      const float bh_r = bhh1[jn], bh_z = bhh1[H + jn], bh_n = bhh1[2 * H + jn];
      const float* hold = hbuf + (size_t)p * PH + LH;
      float* hnf = hbuf + (size_t)(p ^ 1) * PH + LH;
      u16*   hnb = hbf  + (size_t)(p ^ 1) * PH + LH;
#pragma unroll
      for (int v = 0; v < 8; ++v) {
        int m = (wave << 4) + ((lane >> 4) << 3) + v;
        float ir = ai[0][v] + bi_r, iz = ai[1][v] + bi_z, in_ = ai[2][v] + bi_n;
        float hr = ah[0][v] + bh_r, hz = ah[1][v] + bh_z, hn = ah[2][v] + bh_n;
        float r = sigm(ir + hr), z = sigm(iz + hz);
        float n = tanhf(in_ + r * hn);
        float ho = hold[(size_t)m * H + jn];
        float hv = (1.0f - z) * n + z * ho;
        hnf[(size_t)m * H + jn] = hv;
        hnb[(size_t)m * H + jn] = f2bf(hv);
      }
    }
    grid_sync(bar);
    // ---------------- FC head (produces y, fed back as next input) ----------------
    if (s >= SEQ - 1) {
      const float* top = hbuf + (size_t)(p ^ 1) * PH + LH;   // layer-1 new h
#pragma unroll 1
      for (int rr = 0; rr < 2; ++rr) {
        int b = wg * 2 + rr;
        const float* row = top + (size_t)b * H;
        float part = 0.0f;
#pragma unroll
        for (int e = 0; e < 4; ++e) part += row[tid * 4 + e] * fcw[tid * 4 + e];
        red[tid] = part;
        __syncthreads();
        for (int off = TPB / 2; off > 0; off >>= 1) {
          if (tid < off) red[tid] += red[tid + off];
          __syncthreads();
        }
        if (tid == 0) {
          float y = red[0] + fcb[0];
          ybuf[b] = y;
          s_ysum[rr] += y;
        }
        __syncthreads();
      }
      grid_sync(bar);                         // ybuf visible to all WGs for next step
    }
    p ^= 1;
  }

  // ---------------- outputs ----------------
  if (tid == 0) {
    out_mean[2 * wg]     = s_ysum[0] / (float)F;
    out_mean[2 * wg + 1] = s_ysum[1] / (float)F;
  }
  const float* src = hbuf + (size_t)p * PH;   // h_final [L][B][H]
  const size_t gsz = (size_t)NWG * TPB;
  for (size_t i = (size_t)wg * TPB + tid; i < PH; i += gsz) out_h[i] = src[i];
}

extern "C" void kernel_launch(void* const* d_in, const int* in_sizes, int n_in,
                              void* d_out, int out_size, void* d_ws, size_t ws_size,
                              hipStream_t stream) {
  const float* x    = (const float*)d_in[0];
  const float* h0   = (const float*)d_in[1];
  const float* wih0 = (const float*)d_in[2];
  const float* whh0 = (const float*)d_in[3];
  const float* bih0 = (const float*)d_in[4];
  const float* bhh0 = (const float*)d_in[5];
  const float* wih1 = (const float*)d_in[6];
  const float* whh1 = (const float*)d_in[7];
  const float* bih1 = (const float*)d_in[8];
  const float* bhh1 = (const float*)d_in[9];
  const float* fcw  = (const float*)d_in[10];
  const float* fcb  = (const float*)d_in[11];
  const int*   flen = (const int*)d_in[12];

  // workspace layout (~22.3 MB): [barrier 256B][bf16 packed weights 18.9MB]
  //                              [h fp32 2x2xBxH 2MB][h bf16 1MB][ybuf 512B]
  char* ws = (char*)d_ws;
  int*  bar   = (int*)ws;
  u16*  wpack = (u16*)(ws + 256);
  size_t wbytes = (size_t)9 * H * H * sizeof(u16);
  float* hbuf = (float*)(ws + 256 + wbytes);
  size_t hfb  = (size_t)2 * 2 * B * H * sizeof(float);
  u16*  hbf   = (u16*)((char*)hbuf + hfb);
  size_t hbb  = hfb / 2;
  float* ybuf = (float*)((char*)hbf + hbb);
  float* out_mean = (float*)d_out;
  float* out_h    = out_mean + B;

  (void)hipMemsetAsync(d_ws, 0, 256, stream); // deterministic barrier init each call
  pack_weights<<<4096, TPB, 0, stream>>>(whh0, wih1, whh1, h0, wpack, hbuf, hbf);
  gru_persistent<<<NWG, TPB, 0, stream>>>(x, wih0, bih0, bhh0, bih1, bhh1, fcw, fcb, flen,
                                          wpack, hbuf, hbf, ybuf, bar, out_mean, out_h);
}
